// Decoder_12919261626871
// MI455X (gfx1250) — compile-verified, matrix operand order checked by reading, not run on hardware
//
#include <hip/hip_runtime.h>
#include <hip/hip_bf16.h>
#include <math.h>

// ---------------- problem constants ----------------
#define HID     512
#define VOCAB   50000
#define STEPS   31          // CH_SEQ_LEN - 1
#define BATCH   64
#define SRC_LEN 50
#define GATES   2048        // 4*HID
#define KCAT    1024        // 2*HID  (x | h concatenated K)

typedef __attribute__((ext_vector_type(16))) __bf16 bf16x16;
typedef __attribute__((ext_vector_type(8)))  float  f32x8;

// Load one wave-lane's 16 bf16 values for a 16x32 bf16 WMMA operand:
// halves [k0 .. k0+7] and [k0+16 .. k0+23] relative to `p` (p already includes k0).
static __device__ inline bf16x16 load_op16(const __hip_bfloat16* p) {
    union { bf16x16 v; uint4 q[2]; } u;
    u.q[0] = *reinterpret_cast<const uint4*>(p);        // K = k0 .. k0+7
    u.q[1] = *reinterpret_cast<const uint4*>(p + 16);   // K = k0+16 .. k0+23
    return u.v;
}

// ---------------- prologue: fp32 -> bf16 conversions ----------------
__global__ void f2bf_kernel(const float* __restrict__ src,
                            __hip_bfloat16* __restrict__ dst, int n) {
    int i = blockIdx.x * blockDim.x + threadIdx.x;
    if (i < n) dst[i] = __float2bfloat16(src[i]);
}

// Wcat[n, k] = k < HID ? w_ih[n,k] : w_hh[n,k-HID]   (row-major [GATES, KCAT], bf16)
__global__ void build_wcat_kernel(const float* __restrict__ w_ih,
                                  const float* __restrict__ w_hh,
                                  __hip_bfloat16* __restrict__ wcat) {
    int i = blockIdx.x * blockDim.x + threadIdx.x;      // 0 .. GATES*KCAT-1
    int n = i >> 10;
    int k = i & (KCAT - 1);
    float v = (k < HID) ? w_ih[(size_t)n * HID + k]
                        : w_hh[(size_t)n * HID + (k - HID)];
    wcat[i] = __float2bfloat16(v);
}

// ---------------- gates GEMM: [64,2048] = Acat[64,1024] @ Wcat^T ----------------
// One wave per N-tile, all 4 M-tiles blocked in registers (B loaded once per k-step).
// K loop split at the static x|h boundary so each half has plain strided addressing
// (no per-iteration pointer selects).
__global__ __launch_bounds__(256)
void gates_gemm_kernel(const int* __restrict__ tok,
                       const __hip_bfloat16* __restrict__ embed_bf,
                       const __hip_bfloat16* __restrict__ h_carry_bf,
                       const __hip_bfloat16* __restrict__ wcat,
                       float* __restrict__ gates) {
    const int wid  = threadIdx.x >> 5;
    const int lane = threadIdx.x & 31;
    const int nTile = blockIdx.x * 8 + wid;    // 0..127
    const int nBase = nTile * 16;
    const int l15 = lane & 15;
    const int k0  = (lane & 16) ? 8 : 0;

    const __hip_bfloat16* pa_emb[4];
    const __hip_bfloat16* pa_h[4];
    #pragma unroll
    for (int m = 0; m < 4; ++m) {
        int b = m * 16 + l15;
        pa_emb[m] = embed_bf + (size_t)tok[b] * HID + k0;
        pa_h[m]   = h_carry_bf + (size_t)b * HID + k0;
    }
    const __hip_bfloat16* pb = wcat + (size_t)(nBase + l15) * KCAT + k0;

    f32x8 acc[4];
    #pragma unroll
    for (int m = 0; m < 4; ++m)
        acc[m] = (f32x8){0.f,0.f,0.f,0.f,0.f,0.f,0.f,0.f};

    // first half of K: x = embed_bf[tok[b]]
    #pragma unroll 4
    for (int kk = 0; kk < HID; kk += 32) {
        bf16x16 bm = load_op16(pb + kk);
        #pragma unroll
        for (int m = 0; m < 4; ++m) {
            bf16x16 a = load_op16(pa_emb[m] + kk);
            acc[m] = __builtin_amdgcn_wmma_f32_16x16x32_bf16(
                         false, a, false, bm, (short)0, acc[m], false, false);
        }
    }
    // second half of K: h carry
    #pragma unroll 4
    for (int kk = 0; kk < HID; kk += 32) {
        bf16x16 bm = load_op16(pb + HID + kk);
        #pragma unroll
        for (int m = 0; m < 4; ++m) {
            bf16x16 a = load_op16(pa_h[m] + kk);
            acc[m] = __builtin_amdgcn_wmma_f32_16x16x32_bf16(
                         false, a, false, bm, (short)0, acc[m], false, false);
        }
    }

    const int mrow_off = (lane >> 4) << 3;
    #pragma unroll
    for (int m = 0; m < 4; ++m) {
        #pragma unroll
        for (int r = 0; r < 8; ++r)
            gates[(size_t)(m * 16 + mrow_off + r) * GATES + nBase + l15] = acc[m][r];
    }
}

// ---------------- LSTM cell epilogue ----------------
__global__ void cell_kernel(const float* __restrict__ gates,
                            const float* __restrict__ b_ih,
                            const float* __restrict__ b_hh,
                            const float* __restrict__ c_prev,
                            float* __restrict__ h_rnn_f,
                            __hip_bfloat16* __restrict__ h_rnn_bf,
                            float* __restrict__ c_rnn_f) {
    int idx = blockIdx.x * blockDim.x + threadIdx.x;    // 0 .. BATCH*HID-1
    int b = idx >> 9;
    int j = idx & (HID - 1);
    const float* g = gates + (size_t)b * GATES;
    float gi = g[j]            + b_ih[j]            + b_hh[j];
    float gf = g[HID + j]      + b_ih[HID + j]      + b_hh[HID + j];
    float gg = g[2 * HID + j]  + b_ih[2 * HID + j]  + b_hh[2 * HID + j];
    float go = g[3 * HID + j]  + b_ih[3 * HID + j]  + b_hh[3 * HID + j];
    float i_ = 1.f / (1.f + expf(-gi));
    float f_ = 1.f / (1.f + expf(-gf));
    float o_ = 1.f / (1.f + expf(-go));
    float gt = tanhf(gg);
    float c  = f_ * c_prev[idx] + i_ * gt;
    float h  = o_ * tanhf(c);
    c_rnn_f[idx]  = c;
    h_rnn_f[idx]  = h;
    h_rnn_bf[idx] = __float2bfloat16(h);
}

// ---------------- residual attention (for h and c) ----------------
// blockIdx.x = batch, blockIdx.y = 0 (h path -> bf16 carry) / 1 (c path -> f32 carry)
__global__ __launch_bounds__(128)
void attn_kernel(const float* __restrict__ enc,        // [S, B, H]
                 const float* __restrict__ h_rnn_f,
                 const float* __restrict__ c_rnn_f,
                 __hip_bfloat16* __restrict__ h_carry_bf,
                 float* __restrict__ c_state) {
    const int b   = blockIdx.x;
    const int tid = threadIdx.x;
    const float* q = ((blockIdx.y == 0) ? h_rnn_f : c_rnn_f) + (size_t)b * HID;

    __shared__ float red[128];
    __shared__ float sc[SRC_LEN];

    float qv[4];
    #pragma unroll
    for (int i = 0; i < 4; ++i) qv[i] = q[tid + i * 128];

    const float scale = 0.044194173824159216f;   // 1/sqrt(512)
    for (int s = 0; s < SRC_LEN; ++s) {
        const float* er = enc + ((size_t)s * BATCH + b) * HID;
        float p = 0.f;
        #pragma unroll
        for (int i = 0; i < 4; ++i) p += er[tid + i * 128] * qv[i];
        red[tid] = p;
        __syncthreads();
        for (int off = 64; off > 0; off >>= 1) {
            if (tid < off) red[tid] += red[tid + off];
            __syncthreads();
        }
        if (tid == 0) sc[s] = red[0] * scale;
        __syncthreads();
    }

    float m = -3.4e38f;
    for (int s = 0; s < SRC_LEN; ++s) m = fmaxf(m, sc[s]);
    float denom = 0.f;
    for (int s = 0; s < SRC_LEN; ++s) denom += expf(sc[s] - m);

    float acc[4] = {0.f, 0.f, 0.f, 0.f};
    for (int s = 0; s < SRC_LEN; ++s) {
        float w = expf(sc[s] - m);
        const float* er = enc + ((size_t)s * BATCH + b) * HID;
        #pragma unroll
        for (int i = 0; i < 4; ++i) acc[i] += w * er[tid + i * 128];
    }
    #pragma unroll
    for (int i = 0; i < 4; ++i) {
        float o = acc[i] / denom + qv[i];
        size_t e = (size_t)b * HID + tid + i * 128;
        if (blockIdx.y == 0) h_carry_bf[e] = __float2bfloat16(o);
        else                 c_state[e]    = o;
    }
}

// ---------------- logits GEMM: [64, 50000] = h_rnn_bf @ embed_bf^T + bias ----------------
// One wave per N-tile (3125 total); all 4 M-tiles blocked in registers so each
// embedding row is read from L2 exactly once per step (B-reuse 4x).
__global__ __launch_bounds__(256)
void logits_gemm_kernel(const __hip_bfloat16* __restrict__ h_rnn_bf,
                        const __hip_bfloat16* __restrict__ embed_bf,
                        const float* __restrict__ bias,
                        float* __restrict__ out) {     // [BATCH, VOCAB] slab for this step
    const int wid  = threadIdx.x >> 5;
    const int lane = threadIdx.x & 31;
    const int nTile = blockIdx.x * 8 + wid;
    if (nTile >= VOCAB / 16) return;                   // wave-uniform
    const int nBase = nTile * 16;
    const int l15 = lane & 15;
    const int k0  = (lane & 16) ? 8 : 0;

    const __hip_bfloat16* pa0 = h_rnn_bf + (size_t)l15 * HID + k0;  // A tile m=0
    const __hip_bfloat16* pb  = embed_bf + (size_t)(nBase + l15) * HID + k0;
    __builtin_prefetch((const void*)pb, 0, 1);

    f32x8 acc[4];
    #pragma unroll
    for (int m = 0; m < 4; ++m)
        acc[m] = (f32x8){0.f,0.f,0.f,0.f,0.f,0.f,0.f,0.f};

    #pragma unroll 4
    for (int kk = 0; kk < HID; kk += 32) {
        bf16x16 bm = load_op16(pb + kk);
        #pragma unroll
        for (int m = 0; m < 4; ++m) {
            bf16x16 a = load_op16(pa0 + (size_t)m * 16 * HID + kk);
            acc[m] = __builtin_amdgcn_wmma_f32_16x16x32_bf16(
                         false, a, false, bm, (short)0, acc[m], false, false);
        }
    }
    const float bv = bias[nBase + l15];
    const int mrow_off = (lane >> 4) << 3;
    #pragma unroll
    for (int m = 0; m < 4; ++m) {
        #pragma unroll
        for (int r = 0; r < 8; ++r)
            out[(size_t)(m * 16 + mrow_off + r) * VOCAB + nBase + l15]
                = acc[m][r] + bv;
    }
}

// ---------------- per-row argmax over VOCAB ----------------
__global__ __launch_bounds__(256)
void argmax_kernel(const float* __restrict__ logits,   // [BATCH, VOCAB] slab
                   int* __restrict__ tok,
                   float* __restrict__ word_out,       // [BATCH, STEPS] region of d_out
                   int t) {
    const int b   = blockIdx.x;
    const int tid = threadIdx.x;
    const float* row = logits + (size_t)b * VOCAB;
    float best = -3.4e38f; int bidx = 0x7fffffff;
    for (int i = tid; i < VOCAB; i += 256) {
        float v = row[i];
        if (v > best || (v == best && i < bidx)) { best = v; bidx = i; }
    }
    __shared__ float sv[256];
    __shared__ int   si[256];
    sv[tid] = best; si[tid] = bidx;
    __syncthreads();
    for (int off = 128; off > 0; off >>= 1) {
        if (tid < off) {
            if (sv[tid + off] > sv[tid] ||
                (sv[tid + off] == sv[tid] && si[tid + off] < si[tid])) {
                sv[tid] = sv[tid + off]; si[tid] = si[tid + off];
            }
        }
        __syncthreads();
    }
    if (tid == 0) {
        tok[b] = si[0];
        word_out[(size_t)b * STEPS + t] = (float)si[0];
    }
}

// ---------------- host orchestration ----------------
extern "C" void kernel_launch(void* const* d_in, const int* in_sizes, int n_in,
                              void* d_out, int out_size, void* d_ws, size_t ws_size,
                              hipStream_t stream) {
    const float* embed_w   = (const float*)d_in[0];   // [VOCAB, HID]
    const float* fc_bias   = (const float*)d_in[1];   // [VOCAB]
    const float* w_ih      = (const float*)d_in[2];   // [4H, H]
    const float* w_hh      = (const float*)d_in[3];   // [4H, H]
    const float* b_ih      = (const float*)d_in[4];   // [4H]
    const float* b_hh      = (const float*)d_in[5];   // [4H]
    const float* enc       = (const float*)d_in[6];   // [S, B, H]
    const float* h0        = (const float*)d_in[7];   // [B, H]
    const float* c0        = (const float*)d_in[8];   // [B, H]
    const int*   init_tok  = (const int*)d_in[9];     // [B]

    float* out = (float*)d_out;
    float* word_out = out + (size_t)STEPS * BATCH * VOCAB;

    // scratch carve-out (256B aligned)
    char* ws = (char*)d_ws;
    size_t off = 0;
    auto carve = [&](size_t bytes) -> char* {
        char* p = ws + off;
        off = (off + bytes + 255) & ~(size_t)255;
        return p;
    };
    __hip_bfloat16* embed_bf   = (__hip_bfloat16*)carve((size_t)VOCAB * HID * 2);
    __hip_bfloat16* wcat       = (__hip_bfloat16*)carve((size_t)GATES * KCAT * 2);
    __hip_bfloat16* h_carry_bf = (__hip_bfloat16*)carve((size_t)BATCH * HID * 2);
    float*          gates      = (float*)carve((size_t)BATCH * GATES * 4);
    float*          h_rnn_f    = (float*)carve((size_t)BATCH * HID * 4);
    __hip_bfloat16* h_rnn_bf   = (__hip_bfloat16*)carve((size_t)BATCH * HID * 2);
    float*          c_rnn_f    = (float*)carve((size_t)BATCH * HID * 4);
    float*          c_state    = (float*)carve((size_t)BATCH * HID * 4);
    int*            tok        = (int*)carve((size_t)BATCH * 4);
    (void)ws_size; (void)in_sizes; (void)n_in; (void)out_size;

    // prologue: bf16 copies (deterministic, re-done every call)
    {
        int n = VOCAB * HID;
        f2bf_kernel<<<(n + 255) / 256, 256, 0, stream>>>(embed_w, embed_bf, n);
    }
    build_wcat_kernel<<<(GATES * KCAT) / 256, 256, 0, stream>>>(w_ih, w_hh, wcat);
    {
        int n = BATCH * HID;
        f2bf_kernel<<<(n + 255) / 256, 256, 0, stream>>>(h0, h_carry_bf, n);
    }

    const int logits_blocks = (VOCAB / 16 + 7) / 8;               // 391
    const int gates_blocks  = (GATES / 16) / 8;                   // 16

    const int*   tok_src = init_tok;
    const float* c_src   = c0;
    for (int t = 0; t < STEPS; ++t) {
        gates_gemm_kernel<<<gates_blocks, 256, 0, stream>>>(tok_src, embed_bf,
                                                            h_carry_bf, wcat, gates);
        cell_kernel<<<(BATCH * HID) / 256, 256, 0, stream>>>(gates, b_ih, b_hh, c_src,
                                                             h_rnn_f, h_rnn_bf, c_rnn_f);
        attn_kernel<<<dim3(BATCH, 2), 128, 0, stream>>>(enc, h_rnn_f, c_rnn_f,
                                                        h_carry_bf, c_state);
        float* slab = out + (size_t)t * BATCH * VOCAB;
        logits_gemm_kernel<<<logits_blocks, 256, 0, stream>>>(h_rnn_bf, embed_bf,
                                                              fc_bias, slab);
        argmax_kernel<<<BATCH, 256, 0, stream>>>(slab, tok, word_out, t);
        tok_src = tok;
        c_src   = c_state;
    }
}
